// AutoEncoder_32143535244000
// MI455X (gfx1250) — matrix operand order repaired and checked by reading, not compiled, not run on hardware
//
#include <hip/hip_runtime.h>
#include <hip/hip_bf16.h>
#include <math.h>

typedef __bf16 bf16_t;
typedef __attribute__((ext_vector_type(16))) __bf16 v16bf;
typedef __attribute__((ext_vector_type(8)))  __bf16 v8bf;
typedef __attribute__((ext_vector_type(8)))  float   v8f;

#define MBLK 2   // M tiles per wave   (M must be % 32 == 0)
#define NBLK 4   // N tiles per wave   (Npad must be % 64 == 0)

// ---------------------------------------------------------------------------
// Generic bf16 WMMA GEMM: C[M,N] = act(A[M,K] @ Bpacked + bias)
//   A: bf16 row-major [M,K], K % 32 == 0, M % 32 == 0
//   Bp: packed per 16x16-N x 32-K tile in the exact WMMA B VGPR layout:
//       Bp[(((nt*KC)+kc)*32 + lane)*16 + e]  where n = nt*16 + lane%16,
//                                                 k = kc*32 + (lane/16)*8 + (e%8) + (e/8)*16
//   Each wave computes a 32x64 C tile: 8 WMMAs per 32-K chunk,
//   2 A-tiles x 4 B-tiles -> every B load feeds 2 WMMAs, every A load 4.
//   C : f32, row stride ldc, only n < Nclip written
//   Cb: optional bf16 mirror (same indexing), may be null
//   swapTB: row remap m=(t*64+b) -> row = b*52 + t   (for [T,B]->[B,T] output)
// ---------------------------------------------------------------------------
__global__ __launch_bounds__(256)
void gemm_bf16_wmma(const bf16_t* __restrict__ A, const bf16_t* __restrict__ Bp,
                    const float* __restrict__ bias,
                    float* __restrict__ C, bf16_t* __restrict__ Cb,
                    int M, int Npad, int K, int Nclip, int ldc,
                    int relu, int swapTB)
{
    const int gw   = (blockIdx.x * blockDim.x + threadIdx.x) >> 5;
    const int lane = threadIdx.x & 31;
    const int MT2 = M >> 5;                   // 32-row M groups, exact
    const int KC  = K >> 5;
    const int NG  = Npad >> 6;                // 64-col N groups, exact
    if (gw >= MT2 * NG) return;
    const int mg = gw % MT2;
    const int ng = gw / MT2;

    const int half = lane >> 4;
    const int mrow = mg * 32 + (lane & 15);   // A row for m-tile 0; +16 for m-tile 1
    const bf16_t* __restrict__ arow0 = A + (size_t)mrow * K;
    const bf16_t* __restrict__ arow1 = arow0 + (size_t)16 * K;

    // B pointer for (nt = ng*NBLK, kc = 0), this lane's 16-element chunk.
    const v16bf* __restrict__ bptr =
        (const v16bf*)Bp + ((size_t)(ng * NBLK) * KC * 32 + lane);
    const size_t bstride_nt = (size_t)KC * 32;

    v8f acc[MBLK][NBLK];
    #pragma unroll
    for (int mi = 0; mi < MBLK; ++mi)
        #pragma unroll
        for (int j = 0; j < NBLK; ++j)
            #pragma unroll
            for (int i = 0; i < 8; ++i) acc[mi][j][i] = 0.0f;

    for (int kc = 0; kc < KC; ++kc) {
        const int k0 = (kc << 5) + (half << 3);
        v8bf lo0 = *(const v8bf*)(arow0 + k0);
        v8bf hi0 = *(const v8bf*)(arow0 + k0 + 16);
        v8bf lo1 = *(const v8bf*)(arow1 + k0);
        v8bf hi1 = *(const v8bf*)(arow1 + k0 + 16);
        v16bf a0, a1;
        #pragma unroll
        for (int i = 0; i < 8; ++i) {
            a0[i] = lo0[i]; a0[i + 8] = hi0[i];
            a1[i] = lo1[i]; a1[i + 8] = hi1[i];
        }

        const v16bf* bk = bptr + (size_t)kc * 32;
        #pragma unroll
        for (int j = 0; j < NBLK; ++j) {
            v16bf bm = bk[(size_t)j * bstride_nt];
            acc[0][j] = __builtin_amdgcn_wmma_f32_16x16x32_bf16(
                false, a0, false, bm, (short)0, acc[0][j], false, false);
            acc[1][j] = __builtin_amdgcn_wmma_f32_16x16x32_bf16(
                false, a1, false, bm, (short)0, acc[1][j], false, false);
        }
    }

    // Epilogue.  C layout: VGPR i, lane l -> m = mbase + (l/16)*8 + i, n = nt*16 + l%16
    const int nl = lane & 15;
    #pragma unroll
    for (int mi = 0; mi < MBLK; ++mi) {
        const int m0 = mg * 32 + mi * 16 + (half << 3);
        #pragma unroll
        for (int j = 0; j < NBLK; ++j) {
            const int n = (ng * NBLK + j) * 16 + nl;
            if (n >= Nclip) continue;
            const float bv = bias ? bias[n] : 0.0f;
            #pragma unroll
            for (int i = 0; i < 8; ++i) {
                const int m = m0 + i;
                float v = acc[mi][j][i] + bv;
                if (relu) v = fmaxf(v, 0.0f);
                const int row = swapTB ? ((m & 63) * 52 + (m >> 6)) : m;
                const size_t idx = (size_t)row * ldc + n;
                if (C)  C[idx]  = v;
                if (Cb) Cb[idx] = (bf16_t)v;
            }
        }
    }
}

// ---- pack f32 weight [K, Norig] -> bf16 WMMA-B layout, N padded to Npad ----
__global__ __launch_bounds__(256)
void pack_b_kernel(const float* __restrict__ W, bf16_t* __restrict__ Bp,
                   int K, int Norig, int Npad)
{
    const int KC = K >> 5;
    const int total = (Npad >> 4) * KC * 512;
    int p = blockIdx.x * 256 + threadIdx.x;
    if (p >= total) return;
    const int e    = p & 15;
    const int lane = (p >> 4) & 31;
    const int kc   = (p >> 9) % KC;
    const int nt   = p / (KC << 9);
    const int k = (kc << 5) + ((lane >> 4) << 3) + (e & 7) + ((e >> 3) << 4);
    const int n = (nt << 4) + (lane & 15);
    const float v = (n < Norig) ? W[(size_t)k * Norig + n] : 0.0f;
    Bp[p] = (bf16_t)v;
}

__global__ __launch_bounds__(256)
void f32_to_bf16_kernel(const float* __restrict__ x, bf16_t* __restrict__ y, int n)
{
    int i = blockIdx.x * 256 + threadIdx.x;
    if (i < n) y[i] = (bf16_t)x[i];
}

// teacher-forced decoder inputs: xs[t,b,:] = emb[token(t,b)], bf16
__global__ __launch_bounds__(256)
void embed_kernel(const int* __restrict__ label, const float* __restrict__ emb,
                  bf16_t* __restrict__ xs)
{
    int i = blockIdx.x * 256 + threadIdx.x;        // [T=52][B=64][E=256]
    if (i >= 52 * 64 * 256) return;
    const int e = i & 255;
    const int b = (i >> 8) & 63;
    const int t = i >> 14;
    const int tok = (t == 0) ? 3 : label[b * 52 + (t - 1)];
    xs[i] = (bf16_t)emb[tok * 256 + e];
}

__global__ __launch_bounds__(256)
void zero_h_kernel(float* __restrict__ h, bf16_t* __restrict__ hb)
{
    int i = blockIdx.x * 256 + threadIdx.x;
    if (i < 64 * 512) { h[i] = 0.0f; hb[i] = (bf16_t)0.0f; }
}

// Bahdanau attention + GRU input assembly for one step.
// block b (64 blocks, 256 threads): score -> softmax over L=64 -> ctx,
// then xb[b] = concat(ctx, x_emb) in bf16.
__global__ __launch_bounds__(256)
void attn_kernel(const float* __restrict__ featW1, const float* __restrict__ feat,
                 const float* __restrict__ hW2, const float* __restrict__ V_W,
                 const float* __restrict__ V_b, const bf16_t* __restrict__ xs_t,
                 bf16_t* __restrict__ xb)
{
    const int b    = blockIdx.x;
    const int tid  = threadIdx.x;
    const int wv   = tid >> 5;
    const int lane = tid & 31;
    __shared__ float sc[64];
    __shared__ float aw[64];

    // scores: wave wv handles l = wv*8 .. wv*8+7
    #pragma unroll
    for (int li = 0; li < 8; ++li) {
        const int l = wv * 8 + li;
        const float* fw = featW1 + ((size_t)(b * 64 + l)) * 512;
        const float* hw = hW2 + (size_t)b * 512;
        float s = 0.0f;
        for (int k = lane; k < 512; k += 32)
            s += tanhf(fw[k] + hw[k]) * V_W[k];
        #pragma unroll
        for (int off = 16; off > 0; off >>= 1) s += __shfl_xor(s, off);
        if (lane == 0) sc[l] = s + V_b[0];
    }
    __syncthreads();

    // softmax over 64 (first wave, 2 values per lane)
    if (wv == 0) {
        float a0 = sc[lane], a1 = sc[lane + 32];
        float mx = fmaxf(a0, a1);
        #pragma unroll
        for (int off = 16; off > 0; off >>= 1) mx = fmaxf(mx, __shfl_xor(mx, off));
        float e0 = expf(a0 - mx), e1 = expf(a1 - mx);
        float sm = e0 + e1;
        #pragma unroll
        for (int off = 16; off > 0; off >>= 1) sm += __shfl_xor(sm, off);
        aw[lane] = e0 / sm; aw[lane + 32] = e1 / sm;
    }
    __syncthreads();

    // ctx[e] = sum_l aw[l] * feat[b,l,e]   (e = tid, 256 threads)
    float acc = 0.0f;
    const float* fb = feat + (size_t)b * 64 * 256;
    for (int l = 0; l < 64; ++l) acc += aw[l] * fb[l * 256 + tid];
    xb[b * 512 + tid]       = (bf16_t)acc;
    xb[b * 512 + 256 + tid] = xs_t[b * 256 + tid];
}

// GRU gate fusion (keras v2 order z,r,h, reset_after=True)
__global__ __launch_bounds__(256)
void gru_gates_kernel(const float* __restrict__ gx, const float* __restrict__ gh,
                      float* __restrict__ h, bf16_t* __restrict__ hb,
                      bf16_t* __restrict__ Hb_t)
{
    int i = blockIdx.x * 256 + threadIdx.x;
    if (i >= 64 * 512) return;
    const int b = i >> 9, u = i & 511;
    const float* gxr = gx + (size_t)b * 1536;
    const float* ghr = gh + (size_t)b * 1536;
    const float xz = gxr[u], xr = gxr[512 + u], xh = gxr[1024 + u];
    const float hz = ghr[u], hr = ghr[512 + u], hh = ghr[1024 + u];
    const float z = 1.0f / (1.0f + expf(-(xz + hz)));
    const float r = 1.0f / (1.0f + expf(-(xr + hr)));
    const float nn = tanhf(xh + r * hh);
    const float hn = z * h[i] + (1.0f - z) * nn;
    h[i] = hn;
    hb[i] = (bf16_t)hn;
    Hb_t[i] = (bf16_t)hn;
}

__global__ __launch_bounds__(256)
void copy_labels_kernel(const int* __restrict__ label, int* __restrict__ out)
{
    int i = blockIdx.x * 256 + threadIdx.x;
    if (i < 64 * 52) out[i] = label[i];
}

// ---------------------------------------------------------------------------
static inline int ceildiv(int a, int b) { return (a + b - 1) / b; }
static inline int gemm_grid(int M, int Npad) {
    int waves = (M / 32) * (Npad / 64);       // M % 32 == 0, Npad % 64 == 0
    return ceildiv(waves, 8);                 // 8 waves / 256-thread block
}

extern "C" void kernel_launch(void* const* d_in, const int* in_sizes, int n_in,
                              void* d_out, int out_size, void* d_ws, size_t ws_size,
                              hipStream_t stream)
{
    const float* data    = (const float*)d_in[0];
    const int*   label   = (const int*)  d_in[1];
    const float* dense_W = (const float*)d_in[2];
    const float* dense_b = (const float*)d_in[3];
    const float* emb     = (const float*)d_in[4];
    const float* gru_k   = (const float*)d_in[5];
    const float* gru_rk  = (const float*)d_in[6];
    const float* gru_b   = (const float*)d_in[7];
    const float* fc1_W   = (const float*)d_in[8];
    const float* fc1_b   = (const float*)d_in[9];
    const float* fc2_W   = (const float*)d_in[10];
    const float* fc2_b   = (const float*)d_in[11];
    const float* W1      = (const float*)d_in[12];
    const float* W1_b    = (const float*)d_in[13];
    const float* W2      = (const float*)d_in[14];
    const float* W2_b    = (const float*)d_in[15];
    const float* V_W     = (const float*)d_in[16];
    const float* V_b     = (const float*)d_in[17];

    float* outP = (float*)d_out;                             // [B,T,V] f32
    int*   outL = (int*)((float*)d_out + (size_t)64 * 52 * 10001);

    // ---- workspace carve-out ----
    size_t off = 0;
    auto take = [&](size_t bytes) -> void* {
        off = (off + 255) & ~(size_t)255;
        void* p = (char*)d_ws + off;
        off += bytes;
        return p;
    };
    bf16_t* dataB  = (bf16_t*)take((size_t)4096 * 2048 * 2);
    float*  featF  = (float*) take((size_t)4096 * 256 * 4);
    bf16_t* featB  = (bf16_t*)take((size_t)4096 * 256 * 2);
    float*  featW1 = (float*) take((size_t)4096 * 512 * 4);
    bf16_t* dWp    = (bf16_t*)take((size_t)2048 * 256 * 2);
    bf16_t* W1p    = (bf16_t*)take((size_t)256 * 512 * 2);
    bf16_t* W2p    = (bf16_t*)take((size_t)512 * 512 * 2);
    bf16_t* gkp    = (bf16_t*)take((size_t)512 * 1536 * 2);
    bf16_t* grkp   = (bf16_t*)take((size_t)512 * 1536 * 2);
    bf16_t* fc1p   = (bf16_t*)take((size_t)512 * 512 * 2);
    bf16_t* fc2p   = (bf16_t*)take((size_t)512 * 10048 * 2);   // N padded to 64
    bf16_t* xs     = (bf16_t*)take((size_t)52 * 64 * 256 * 2);
    float*  h      = (float*) take((size_t)64 * 512 * 4);
    bf16_t* hb     = (bf16_t*)take((size_t)64 * 512 * 2);
    bf16_t* xb     = (bf16_t*)take((size_t)64 * 512 * 2);
    float*  hW2    = (float*) take((size_t)64 * 512 * 4);
    float*  gx     = (float*) take((size_t)64 * 1536 * 4);
    float*  gh     = (float*) take((size_t)64 * 1536 * 4);
    bf16_t* Hb     = (bf16_t*)take((size_t)52 * 64 * 512 * 2);
    float*  P1     = (float*) take((size_t)3328 * 512 * 4);
    bf16_t* P1b    = (bf16_t*)take((size_t)3328 * 512 * 2);

    // ---- one-time per launch: convert + pack ----
    f32_to_bf16_kernel<<<ceildiv(4096 * 2048, 256), 256, 0, stream>>>(data, dataB, 4096 * 2048);
    pack_b_kernel<<<ceildiv(2048 * 256, 256), 256, 0, stream>>>(dense_W, dWp, 2048, 256, 256);
    pack_b_kernel<<<ceildiv(256 * 512, 256), 256, 0, stream>>>(W1, W1p, 256, 512, 512);
    pack_b_kernel<<<ceildiv(512 * 512, 256), 256, 0, stream>>>(W2, W2p, 512, 512, 512);
    pack_b_kernel<<<ceildiv(512 * 1536, 256), 256, 0, stream>>>(gru_k, gkp, 512, 1536, 1536);
    pack_b_kernel<<<ceildiv(512 * 1536, 256), 256, 0, stream>>>(gru_rk, grkp, 512, 1536, 1536);
    pack_b_kernel<<<ceildiv(512 * 512, 256), 256, 0, stream>>>(fc1_W, fc1p, 512, 512, 512);
    pack_b_kernel<<<ceildiv(512 * 10048, 256), 256, 0, stream>>>(fc2_W, fc2p, 512, 10001, 10048);

    // feat = relu(data @ dense_W + b)  -> f32 + bf16 mirror
    gemm_bf16_wmma<<<gemm_grid(4096, 256), 256, 0, stream>>>(
        dataB, dWp, dense_b, featF, featB, 4096, 256, 2048, 256, 256, 1, 0);
    // featW1 = feat @ W1 + b  (loop-invariant)
    gemm_bf16_wmma<<<gemm_grid(4096, 512), 256, 0, stream>>>(
        featB, W1p, W1_b, featW1, nullptr, 4096, 512, 256, 512, 512, 0, 0);

    embed_kernel<<<ceildiv(52 * 64 * 256, 256), 256, 0, stream>>>(label, emb, xs);
    zero_h_kernel<<<ceildiv(64 * 512, 256), 256, 0, stream>>>(h, hb);

    // ---- sequential decoder: attention + GRU only (output proj hoisted) ----
    for (int t = 0; t < 52; ++t) {
        gemm_bf16_wmma<<<gemm_grid(64, 512), 256, 0, stream>>>(
            hb, W2p, W2_b, hW2, nullptr, 64, 512, 512, 512, 512, 0, 0);
        attn_kernel<<<64, 256, 0, stream>>>(
            featW1, featF, hW2, V_W, V_b, xs + (size_t)t * 64 * 256, xb);
        gemm_bf16_wmma<<<gemm_grid(64, 1536), 256, 0, stream>>>(
            xb, gkp, gru_b, gx, nullptr, 64, 1536, 512, 1536, 1536, 0, 0);
        gemm_bf16_wmma<<<gemm_grid(64, 1536), 256, 0, stream>>>(
            hb, grkp, gru_b + 1536, gh, nullptr, 64, 1536, 512, 1536, 1536, 0, 0);
        gru_gates_kernel<<<ceildiv(64 * 512, 256), 256, 0, stream>>>(
            gx, gh, h, hb, Hb + (size_t)t * 64 * 512);
    }

    // ---- batched output projection over all T*B rows ----
    gemm_bf16_wmma<<<gemm_grid(3328, 512), 256, 0, stream>>>(
        Hb, fc1p, fc1_b, P1, P1b, 3328, 512, 512, 512, 512, 0, 0);
    gemm_bf16_wmma<<<gemm_grid(3328, 10048), 256, 0, stream>>>(
        P1b, fc2p, fc2_b, outP, nullptr, 3328, 10048, 512, 10001, 10001, 0, 1);

    copy_labels_kernel<<<ceildiv(64 * 52, 256), 256, 0, stream>>>(label, outL);
}
